// DualHeadSTAGATELayer_27212912787633
// MI455X (gfx1250) — matrix operand order repaired and checked by reading, not compiled, not run on hardware
//
#include <hip/hip_runtime.h>
#include <hip/hip_bf16.h>
#include <math.h>

typedef __attribute__((ext_vector_type(16))) _Float16 v16h;
typedef __attribute__((ext_vector_type(8)))  float    v8f;

#define NN 100000
#define NE 1600000
#define KDIM 256
#define DDIM 128
#define NEG_SLOPE 0.2f
#define LAM 0.5f
#define EPSV 1e-8f

// ---------------------------------------------------------------------------
// Convert both weight matrices (128x256 f32 each) into one 256x256 f16 matrix
// W16[row][k]: rows 0..127 = W_gat, rows 128..255 = W_fixed (row-major in K).
// ---------------------------------------------------------------------------
__global__ __launch_bounds__(256) void k_convert_w(const float* __restrict__ Wg,
                                                   const float* __restrict__ Wf,
                                                   _Float16* __restrict__ W16) {
    int i = blockIdx.x * 256 + threadIdx.x;       // 0..32767
    W16[i]          = (_Float16)Wg[i];
    W16[32768 + i]  = (_Float16)Wf[i];
}

// ---------------------------------------------------------------------------
// Fused GEMM via WMMA: C(100000 x 256) = X(100000 x 256) * W16^T
// Block = 256 threads (8 waves), each block computes a 16-row x 256-col tile.
// Wave w owns columns [32w, 32w+32) as two 16x16 WMMA tiles.
// Cols 0..127 -> hbuf (h = x@W_gat^T), cols 128..255 -> fbuf (xW = x@W_fixed^T)
// ---------------------------------------------------------------------------
__global__ __launch_bounds__(256) void k_gemm_wmma(const float* __restrict__ x,
                                                   const _Float16* __restrict__ W16,
                                                   float* __restrict__ hbuf,
                                                   float* __restrict__ fbuf) {
    __shared__ _Float16 xs[16 * KDIM];            // 8 KB x-tile, f16
    const int tid  = threadIdx.x;
    const int row0 = blockIdx.x * 16;

    // Stage x tile (16x256) into LDS as f16, coalesced.
    const float* xg = x + (size_t)row0 * KDIM;
#pragma unroll
    for (int i = 0; i < 16; ++i) {
        int e = tid + i * 256;
        xs[e] = (_Float16)xg[e];
    }
    __syncthreads();

    const int wave = tid >> 5;
    const int lane = tid & 31;
    const int m    = lane & 15;
    const int hi   = lane >> 4;                   // 0 | 1 (lane half)
    const int bn0  = wave * 32 + (lane & 15);     // B column, tile 0
    const int bn1  = bn0 + 16;                    // B column, tile 1

    v8f acc0 = {};
    v8f acc1 = {};

#pragma unroll
    for (int kk = 0; kk < KDIM; kk += 32) {
        // A fragment (16x32 f16): lane half selects K 0-15 vs 16-31,
        // 16 consecutive K per lane (matches B), pairs within VGPR.
        v16h a;
        const int kb = kk + hi * 16;
#pragma unroll
        for (int v = 0; v < 8; ++v) {
            int k = kb + 2 * v;
            a[2 * v]     = xs[m * KDIM + k];
            a[2 * v + 1] = xs[m * KDIM + k + 1];
        }
        // B fragment (32x16 f16): lane column = lane&15, lane half selects
        // K 0-15 vs 16-31; 16 consecutive halves (32B aligned).
        const int kh = kk + hi * 16;
        v16h b0 = *(const v16h*)(W16 + (size_t)bn0 * KDIM + kh);
        v16h b1 = *(const v16h*)(W16 + (size_t)bn1 * KDIM + kh);

        acc0 = __builtin_amdgcn_wmma_f32_16x16x32_f16(false, a, false, b0,
                                                      (short)0, acc0, false, false);
        acc1 = __builtin_amdgcn_wmma_f32_16x16x32_f16(false, a, false, b1,
                                                      (short)0, acc1, false, false);
    }

    // D layout: VGPR r -> row m = r + 8*hi, col = lane&15 within tile.
#pragma unroll
    for (int r = 0; r < 8; ++r) {
        int row = row0 + r + hi * 8;
        float* p0 = (bn0 < DDIM) ? (hbuf + (size_t)row * DDIM + bn0)
                                 : (fbuf + (size_t)row * DDIM + (bn0 - DDIM));
        *p0 = acc0[r];
        float* p1 = (bn1 < DDIM) ? (hbuf + (size_t)row * DDIM + bn1)
                                 : (fbuf + (size_t)row * DDIM + (bn1 - DDIM));
        *p1 = acc1[r];
    }
}

// ---------------------------------------------------------------------------
// Per-node attention scores: s_src[n] = h[n].att_src, s_dst[n] = h[n].att_dst
// One wave per node, float4 per lane, wave-shuffle reduction.
// ---------------------------------------------------------------------------
__global__ __launch_bounds__(256) void k_scores(const float* __restrict__ hbuf,
                                                const float* __restrict__ att_src,
                                                const float* __restrict__ att_dst,
                                                float* __restrict__ s_src,
                                                float* __restrict__ s_dst) {
    int wave = threadIdx.x >> 5, lane = threadIdx.x & 31;
    int node = blockIdx.x * 8 + wave;
    if (node >= NN) return;
    float4 h4 = ((const float4*)(hbuf + (size_t)node * DDIM))[lane];
    float4 a4 = ((const float4*)att_src)[lane];
    float4 b4 = ((const float4*)att_dst)[lane];
    float ps = h4.x * a4.x + h4.y * a4.y + h4.z * a4.z + h4.w * a4.w;
    float pd = h4.x * b4.x + h4.y * b4.y + h4.z * b4.z + h4.w * b4.w;
#pragma unroll
    for (int off = 16; off > 0; off >>= 1) {
        ps += __shfl_xor(ps, off, 32);
        pd += __shfl_xor(pd, off, 32);
    }
    if (lane == 0) { s_src[node] = ps; s_dst[node] = pd; }
}

// ---------------------------------------------------------------------------
// Zero the accumulator (d_out) and the segment-sum denominator.
// ---------------------------------------------------------------------------
__global__ __launch_bounds__(256) void k_init(float* __restrict__ out,
                                              float* __restrict__ denom) {
    long long i = (long long)blockIdx.x * 256 + threadIdx.x;
    long long tot = (long long)NN * DDIM;
    if (i < tot)           out[i] = 0.0f;
    else if (i < tot + NN) denom[i - tot] = 0.0f;
}

// ---------------------------------------------------------------------------
// Per-edge scalar pass: alpha = sigmoid(leakyrelu(s_src[src]+s_dst[dst])),
// denom[dst] += alpha (f32 atomic).
// ---------------------------------------------------------------------------
__global__ __launch_bounds__(256) void k_edge_alpha(const long long* __restrict__ ei,
                                                    const float* __restrict__ s_src,
                                                    const float* __restrict__ s_dst,
                                                    float* __restrict__ alpha,
                                                    float* __restrict__ denom) {
    int e = blockIdx.x * 256 + threadIdx.x;
    if (e >= NE) return;
    int s = (int)ei[e];
    int d = (int)ei[NE + e];
    float v = s_src[s] + s_dst[d];
    v = (v >= 0.0f) ? v : NEG_SLOPE * v;
    float a = 1.0f / (1.0f + __expf(-v));
    alpha[e] = a;
    atomicAdd(&denom[d], a);
}

// ---------------------------------------------------------------------------
// Fused edge scatter: out[dst] += lam*alpha_norm*h[src] + (1-lam)*beta*xW[src]
// One wave per edge, float4 per lane (128 cols / 32 lanes).
// ---------------------------------------------------------------------------
__global__ __launch_bounds__(256) void k_edge_scatter(const long long* __restrict__ ei,
                                                      const float* __restrict__ beta,
                                                      const float* __restrict__ alpha,
                                                      const float* __restrict__ denom,
                                                      const float* __restrict__ hbuf,
                                                      const float* __restrict__ fbuf,
                                                      float* __restrict__ out) {
    int wave = threadIdx.x >> 5, lane = threadIdx.x & 31;
    int e = blockIdx.x * 8 + wave;
    if (e >= NE) return;
    int s = (int)ei[e];
    int d = (int)ei[NE + e];
    float wA = LAM * alpha[e] / (denom[d] + EPSV);
    float wF = (1.0f - LAM) * beta[e];
    float4 h4 = ((const float4*)(hbuf + (size_t)s * DDIM))[lane];
    float4 f4 = ((const float4*)(fbuf + (size_t)s * DDIM))[lane];
    float* op = out + (size_t)d * DDIM + lane * 4;
    atomicAdd(op + 0, wA * h4.x + wF * f4.x);
    atomicAdd(op + 1, wA * h4.y + wF * f4.y);
    atomicAdd(op + 2, wA * h4.z + wF * f4.z);
    atomicAdd(op + 3, wA * h4.w + wF * f4.w);
}

// ---------------------------------------------------------------------------
// Finalize in-place: out = elu(out + lam*bias)
// ---------------------------------------------------------------------------
__global__ __launch_bounds__(256) void k_finalize(float* __restrict__ out,
                                                  const float* __restrict__ bias) {
    long long i = (long long)blockIdx.x * 256 + threadIdx.x;
    if (i >= (long long)NN * DDIM) return;
    float v = out[i] + LAM * bias[i & (DDIM - 1)];
    out[i] = (v > 0.0f) ? v : (__expf(v) - 1.0f);
}

extern "C" void kernel_launch(void* const* d_in, const int* in_sizes, int n_in,
                              void* d_out, int out_size, void* d_ws, size_t ws_size,
                              hipStream_t stream) {
    const float*     x       = (const float*)d_in[0];
    const long long* ei      = (const long long*)d_in[1];
    const float*     beta    = (const float*)d_in[2];
    const float*     Wg      = (const float*)d_in[3];
    const float*     att_src = (const float*)d_in[4];
    const float*     att_dst = (const float*)d_in[5];
    const float*     bias    = (const float*)d_in[6];
    const float*     Wf      = (const float*)d_in[7];
    float*           out     = (float*)d_out;

    // Workspace carve-up (all offsets 32B-aligned).
    char* ws = (char*)d_ws;
    size_t off = 0;
    float* hbuf  = (float*)(ws + off); off += (size_t)NN * DDIM * sizeof(float); // 51.2 MB
    float* fbuf  = (float*)(ws + off); off += (size_t)NN * DDIM * sizeof(float); // 51.2 MB
    float* s_src = (float*)(ws + off); off += (size_t)NN * sizeof(float);
    float* s_dst = (float*)(ws + off); off += (size_t)NN * sizeof(float);
    float* alpha = (float*)(ws + off); off += (size_t)NE * sizeof(float);
    float* denom = (float*)(ws + off); off += (size_t)NN * sizeof(float);
    _Float16* W16 = (_Float16*)(ws + off); // 256*256 f16 = 128 KB

    // 1) f32 -> f16 weight conversion (once per launch; tiny)
    k_convert_w<<<128, 256, 0, stream>>>(Wg, Wf, W16);

    // 2) fused dual GEMM via WMMA: N/16 = 6250 row-tiles
    k_gemm_wmma<<<NN / 16, 256, 0, stream>>>(x, W16, hbuf, fbuf);

    // 3) per-node attention scores (factorized attention)
    k_scores<<<NN / 8, 256, 0, stream>>>(hbuf, att_src, att_dst, s_src, s_dst);

    // 4) zero accumulator + denom
    {
        long long tot = (long long)NN * DDIM + NN;
        k_init<<<(unsigned)((tot + 255) / 256), 256, 0, stream>>>(out, denom);
    }

    // 5) per-edge alpha + denominator segment-sum
    k_edge_alpha<<<NE / 256, 256, 0, stream>>>(ei, s_src, s_dst, alpha, denom);

    // 6) fused dual-branch scatter into d_out
    k_edge_scatter<<<NE / 8, 256, 0, stream>>>(ei, beta, alpha, denom, hbuf, fbuf, out);

    // 7) bias + ELU in place
    k_finalize<<<(NN * DDIM) / 256, 256, 0, stream>>>(out, bias);
}
